// ClusteringLayer_53171695125185
// MI455X (gfx1250) — matrix operand order repaired and checked by reading, not compiled, not run on hardware
//
#include <hip/hip_runtime.h>

// ---------------- types for CDNA5 WMMA ----------------
typedef __attribute__((ext_vector_type(16))) __bf16 v16bf;
typedef __attribute__((ext_vector_type(8)))  __bf16 v8bf;
typedef __attribute__((ext_vector_type(8)))  float  v8f;

#define NU   16      // utterances
#define NF   256     // feature dim
#define NT   1500    // duration
#define NTP  1504    // duration padded to multiple of 32 (WMMA K step)
#define NC   64      // clusters kept
#define NCG  66      // clusters + ghosts
#define NCP  80      // padded to 5 WMMA N-tiles
#define TTILES 94    // NTP/16

__device__ __forceinline__ v8f wmma_bf16(v16bf a, v16bf b, v8f c) {
  // D = A(16x32 bf16) * B(32x16 bf16) + C(16x16 f32)
  return __builtin_amdgcn_wmma_f32_16x16x32_bf16(
      /*neg_a=*/false, a, /*neg_b=*/false, b,
      /*c_mod=*/(short)0, c, /*reuse_a=*/false, /*reuse_b=*/false);
}

__device__ __forceinline__ v16bf cat8(v8bf lo, v8bf hi) {
  return __builtin_shufflevector(lo, hi, 0,1,2,3,4,5,6,7,8,9,10,11,12,13,14,15);
}

// ---------------- kernel 0a: pad/convert params, zero asum ----------------
__global__ void k_params(const float* __restrict__ lw, const float* __restrict__ lb,
                         __bf16* __restrict__ wp, float* __restrict__ bp,
                         float* __restrict__ asum) {
  for (int i = threadIdx.x; i < NCP * NF; i += blockDim.x) {
    int c = i >> 8;
    wp[i] = (c < NCG) ? (__bf16)lw[i] : (__bf16)0.f;
  }
  for (int i = threadIdx.x; i < NCP; i += blockDim.x)
    bp[i] = (i < NCG) ? lb[i] : -1e30f;  // padded columns vanish in softmax
  for (int i = threadIdx.x; i < NU * NC; i += blockDim.x)
    asum[i] = 0.f;
}

// ---------------- kernel 0b: x f32[u,f,t] -> bf16 [u,f,tp] and [u,tp,f] ----------------
__global__ void k_convert(const float* __restrict__ x,
                          __bf16* __restrict__ xbf,    // [u, f, NTP]  (B role of GEMM2)
                          __bf16* __restrict__ xtbf) { // [u, NTP, f]  (A role of GEMM1)
  const int row = blockIdx.x;            // u*NF + f
  const int u = row >> 8, f = row & 255;
  const float* src = x + (size_t)row * NT;
  __bf16* drow = xbf + (size_t)row * NTP;
  for (int t = threadIdx.x; t < NTP; t += blockDim.x) {
    float v = (t < NT) ? src[t] : 0.f;
    __bf16 b = (__bf16)v;
    drow[t] = b;
    xtbf[((size_t)u * NTP + t) * NF + f] = b;
  }
}

// ---------------- kernel 1: logits GEMM + fused softmax -> a^T, asum ----------------
// One wave per 16-row t-tile; 5 c-tiles (80 padded cols), K=256 over f.
__global__ void __launch_bounds__(256) k_assign(
    const __bf16* __restrict__ xt,   // [u, NTP, NF]
    const __bf16* __restrict__ wp,   // [NCP, NF]
    const float*  __restrict__ bp,   // [NCP]
    __bf16* __restrict__ at,         // [u, NC, NTP]   a transposed
    float*  __restrict__ asum) {     // [u, NC]
  const int u     = blockIdx.y;
  const int wave  = threadIdx.x >> 5;
  const int lane  = threadIdx.x & 31;
  const int ttile = blockIdx.x * 8 + wave;
  if (ttile >= TTILES) return;
  const int t0 = ttile * 16;
  const int n  = lane & 15;
  const int h  = lane >> 4;
  const int kb = h * 8;

  // A role: row m = lane&15 -> t = t0+m ; K = 16h..16h+15 contiguous
  const __bf16* Arow = xt + ((size_t)u * NTP + t0 + n) * NF;

  float bn[5];
#pragma unroll
  for (int ct = 0; ct < 5; ++ct) bn[ct] = bp[ct * 16 + n];

  const v8f vz = {0.f,0.f,0.f,0.f,0.f,0.f,0.f,0.f};
  v8f acc[5] = {vz, vz, vz, vz, vz};

  for (int f0 = 0; f0 < NF; f0 += 32) {
    v16bf A = *(const v16bf*)(Arow + f0 + 16 * h);
#pragma unroll
    for (int ct = 0; ct < 5; ++ct) {
      // B role: lane = col n -> cluster ct*16+n ; K = 16h..16h+15 contiguous
      v16bf B = *(const v16bf*)(wp + (size_t)(ct * 16 + n) * NF + f0 + 16 * h);
      acc[ct] = wmma_bf16(A, B, acc[ct]);
    }
  }

  // softmax over 80 cols per row; row m = r + 8h lives in 16 lanes (fixed h)
  float psum[4] = {0.f, 0.f, 0.f, 0.f};
#pragma unroll
  for (int r = 0; r < 8; ++r) {
    float v[5];
#pragma unroll
    for (int ct = 0; ct < 5; ++ct) v[ct] = acc[ct][r] + bn[ct];
    float mx = v[0];
#pragma unroll
    for (int ct = 1; ct < 5; ++ct) mx = fmaxf(mx, v[ct]);
    mx = fmaxf(mx, __shfl_xor(mx, 1, 32));
    mx = fmaxf(mx, __shfl_xor(mx, 2, 32));
    mx = fmaxf(mx, __shfl_xor(mx, 4, 32));
    mx = fmaxf(mx, __shfl_xor(mx, 8, 32));
    float e[5], s = 0.f;
#pragma unroll
    for (int ct = 0; ct < 5; ++ct) { e[ct] = __expf(v[ct] - mx); s += e[ct]; }
    s += __shfl_xor(s, 1, 32);
    s += __shfl_xor(s, 2, 32);
    s += __shfl_xor(s, 4, 32);
    s += __shfl_xor(s, 8, 32);
    const float inv = 1.0f / s;
    const int t = t0 + r + 8 * h;
    const bool valid = (t < NT);
#pragma unroll
    for (int ct = 0; ct < 4; ++ct) {   // drop ghost/pad tiles (c >= 64)
      float a = valid ? e[ct] * inv : 0.f;
      at[(size_t)(u * NC + ct * 16 + n) * NTP + t] = (__bf16)a;
      psum[ct] += a;
    }
  }
#pragma unroll
  for (int ct = 0; ct < 4; ++ct)
    atomicAdd(&asum[u * NC + ct * 16 + n], psum[ct]);
}

// ---------------- kernel 2: comp = a^T x^T - asum*centroid, intra-norm ----------------
// Block = (u, 16-cluster tile); 4 waves x 4 f-tiles = full 256-f rows in one block.
__global__ void __launch_bounds__(128) k_aggregate(
    const __bf16* __restrict__ at,    // [u, NC, NTP]
    const __bf16* __restrict__ xb,    // [u, NF, NTP]
    const float*  __restrict__ asum,  // [u, NC]
    const float*  __restrict__ cent,  // [NC, NF]
    float* __restrict__ out) {        // [u, NC*NF]
  __shared__ float rowsq[16];
  const int u    = blockIdx.y;
  const int c0   = blockIdx.x * 16;
  const int wave = threadIdx.x >> 5;
  const int lane = threadIdx.x & 31;
  const int n = lane & 15, h = lane >> 4, kb = h * 8;

  if (threadIdx.x < 16) rowsq[threadIdx.x] = 0.f;
  __syncthreads();

  // A role: row m -> cluster c0+m, K = t
  const __bf16* Arow = at + (size_t)(u * NC + c0 + n) * NTP;
  const __bf16* Bb[4];
  const v8f vz = {0.f,0.f,0.f,0.f,0.f,0.f,0.f,0.f};
  v8f acc[4] = {vz, vz, vz, vz};
#pragma unroll
  for (int i = 0; i < 4; ++i) {
    const int f = (wave * 4 + i) * 16 + n;   // B role: lane = col n -> feature f
    Bb[i] = xb + (size_t)(u * NF + f) * NTP;
  }

  for (int t0 = 0; t0 < NTP; t0 += 32) {
    v16bf A = *(const v16bf*)(Arow + t0 + 16 * h);
#pragma unroll
    for (int i = 0; i < 4; ++i) {
      v16bf B = *(const v16bf*)(Bb[i] + t0 + 16 * h);
      acc[i] = wmma_bf16(A, B, acc[i]);
    }
  }

  // subtract asum*centroid, per-row sumsq (row c = c0 + r + 8h over all 256 f)
  float sq[8];
#pragma unroll
  for (int r = 0; r < 8; ++r) {
    const int c = c0 + r + 8 * h;
    const float as = asum[u * NC + c];
    float s = 0.f;
#pragma unroll
    for (int i = 0; i < 4; ++i) {
      const int f = (wave * 4 + i) * 16 + n;
      float v = acc[i][r] - as * cent[(size_t)c * NF + f];
      acc[i][r] = v;
      s += v * v;
    }
    s += __shfl_xor(s, 1, 32);
    s += __shfl_xor(s, 2, 32);
    s += __shfl_xor(s, 4, 32);
    s += __shfl_xor(s, 8, 32);
    sq[r] = s;
  }
  if (n == 0) {
#pragma unroll
    for (int r = 0; r < 8; ++r) atomicAdd(&rowsq[r + 8 * h], sq[r]);
  }
  __syncthreads();

#pragma unroll
  for (int r = 0; r < 8; ++r) {
    const int c = c0 + r + 8 * h;
    const float scale = 1.0f / fmaxf(sqrtf(rowsq[r + 8 * h]), 1e-12f);
#pragma unroll
    for (int i = 0; i < 4; ++i) {
      const int f = (wave * 4 + i) * 16 + n;
      out[((size_t)u * NC + c) * NF + f] = acc[i][r] * scale;
    }
  }
}

// ---------------- kernel 3: global L2 norm per utterance, in place ----------------
__global__ void __launch_bounds__(256) k_final(float* __restrict__ out) {
  __shared__ float red[256];
  const int u = blockIdx.x;
  float* p = out + (size_t)u * (NC * NF);
  float s = 0.f;
  for (int j = threadIdx.x; j < NC * NF; j += 256) { float v = p[j]; s += v * v; }
  red[threadIdx.x] = s;
  __syncthreads();
  for (int off = 128; off > 0; off >>= 1) {
    if (threadIdx.x < off) red[threadIdx.x] += red[threadIdx.x + off];
    __syncthreads();
  }
  const float scale = 1.0f / fmaxf(sqrtf(red[0]), 1e-12f);
  for (int j = threadIdx.x; j < NC * NF; j += 256) p[j] *= scale;
}

// ---------------- host launcher ----------------
extern "C" void kernel_launch(void* const* d_in, const int* in_sizes, int n_in,
                              void* d_out, int out_size, void* d_ws, size_t ws_size,
                              hipStream_t stream) {
  (void)in_sizes; (void)n_in; (void)out_size; (void)ws_size;
  const float* x    = (const float*)d_in[0];   // [16,256,1500]
  const float* cent = (const float*)d_in[1];   // [64,256]
  const float* lw   = (const float*)d_in[2];   // [66,256]
  const float* lb   = (const float*)d_in[3];   // [66]
  float* out = (float*)d_out;                  // [16, 64*256]

  char* ws = (char*)d_ws;
  size_t o = 0;
  __bf16* xbf  = (__bf16*)(ws + o); o += (size_t)NU * NF  * NTP * 2;  // [u,f,tp]
  __bf16* xtbf = (__bf16*)(ws + o); o += (size_t)NU * NTP * NF  * 2;  // [u,tp,f]
  __bf16* at   = (__bf16*)(ws + o); o += (size_t)NU * NC  * NTP * 2;  // [u,c,tp]
  __bf16* wp   = (__bf16*)(ws + o); o += (size_t)NCP * NF * 2;        // [80,256]
  float*  bp   = (float*) (ws + o); o += 128 * 4;                     // [80] padded
  float*  asum = (float*) (ws + o); o += (size_t)NU * NC * 4;         // [16,64]

  k_params  <<<1, 256, 0, stream>>>(lw, lb, wp, bp, asum);
  k_convert <<<NU * NF, 256, 0, stream>>>(x, xbf, xtbf);
  k_assign  <<<dim3((TTILES + 7) / 8, NU), 256, 0, stream>>>(xtbf, wp, bp, at, asum);
  k_aggregate<<<dim3(NC / 16, NU), 128, 0, stream>>>(at, xbf, asum, cent, out);
  k_final   <<<NU, 256, 0, stream>>>(out);
}
